// TriplaneSelfAttentionLoRAAttnProcessor_90649579749424
// MI455X (gfx1250) — matrix operand order repaired and checked by reading, not compiled, not run on hardware
//
#include <hip/hip_runtime.h>

#define DEVINL __device__ __forceinline__

typedef __attribute__((ext_vector_type(16))) __bf16 v16bf;
typedef __attribute__((ext_vector_type(8)))  __bf16 v8bf;
typedef __attribute__((ext_vector_type(8)))  float  v8f;

constexpr int Sx   = 1024;
constexpr int Cx   = 1024;
constexpr int Hx   = 16;
constexpr int Rx   = 4;
constexpr int ROWS = 3 * Sx;   // 3072 total rows across planes
constexpr int Tx   = 3 * Sx;   // concatenated KV length per plane

// ---------------------------------------------------------------------------
// WMMA helpers (CDNA5 / gfx1250, wave32)
// ---------------------------------------------------------------------------
DEVINL v8f wmma_bf16(v16bf a, v16bf b, v8f c) {
  // D = A(16x32) * B(32x16) + C, f32 accumulate
  return __builtin_amdgcn_wmma_f32_16x16x32_bf16(
      /*neg_a=*/false, a, /*neg_b=*/false, b,
      /*c_mod=*/(short)0, c, /*reuse_a=*/false, /*reuse_b=*/false);
}

// A-operand (16x32 bf16) from row-major src: lane = M (mod 16),
// element e <-> K = e + 16*(lane>=16)   (contiguous, matches B)
DEVINL v16bf load_frag_a(const __bf16* __restrict__ src, int ld, int row0, int col0) {
  const int lane = threadIdx.x & 31;
  const __bf16* base = src + (size_t)(row0 + (lane & 15)) * ld + col0 + ((lane >> 4) << 4);
  v8bf lo = *(const v8bf*)(base);        // K chunk {0..7} (+16 for hi lanes)
  v8bf hi = *(const v8bf*)(base + 8);    // K chunk {8..15} (+16 for hi lanes)
  v16bf f;
#pragma unroll
  for (int e = 0; e < 8; ++e) { f[e] = lo[e]; f[e + 8] = hi[e]; }
  return f;
}

// B-operand (32x16 bf16) where B[k][n] = src[row0+n][col0+k] (src row-major):
// lane = N (mod 16), element e <-> K = e + 16*(lane>=16)  -> 32B contiguous
DEVINL v16bf load_frag_b(const __bf16* __restrict__ src, int ld, int row0, int col0) {
  const int lane = threadIdx.x & 31;
  const __bf16* base = src + (size_t)(row0 + (lane & 15)) * ld + col0 + ((lane >> 4) << 4);
  return *(const v16bf*)(base);
}

// ---------------------------------------------------------------------------
// fp32 -> bf16 conversion (vectorized)
// ---------------------------------------------------------------------------
__global__ __launch_bounds__(256) void cvt_f32_bf16(const float* __restrict__ src,
                                                    __bf16* __restrict__ dst, int n4) {
  int i = blockIdx.x * blockDim.x + threadIdx.x;
  if (i >= n4) return;
  float4 v = ((const float4*)src)[i];
  __bf16* d = dst + (size_t)i * 4;
  d[0] = (__bf16)v.x; d[1] = (__bf16)v.y; d[2] = (__bf16)v.z; d[3] = (__bf16)v.w;
}

// ---------------------------------------------------------------------------
// LoRA down-projection: out[row][r] = sum_c X[row][c] * down[plane][r][c]
// one wave per row (fp32, rank 4)
// ---------------------------------------------------------------------------
__global__ __launch_bounds__(128) void lora_down_k(const float* __restrict__ X,
                                                   const float* __restrict__ dn,
                                                   float* __restrict__ out) {
  const int row  = blockIdx.x * 4 + (threadIdx.x >> 5);
  const int lane = threadIdx.x & 31;
  const int p    = row >> 10;                 // row / Sx
  const float* x  = X + (size_t)row * Cx;
  const float* d0 = dn + (size_t)p * Rx * Cx;
  float a0 = 0.f, a1 = 0.f, a2 = 0.f, a3 = 0.f;
  for (int c = lane; c < Cx; c += 32) {
    float xv = x[c];
    a0 += xv * d0[c];
    a1 += xv * d0[Cx + c];
    a2 += xv * d0[2 * Cx + c];
    a3 += xv * d0[3 * Cx + c];
  }
#pragma unroll
  for (int m = 16; m >= 1; m >>= 1) {
    a0 += __shfl_xor(a0, m, 32);
    a1 += __shfl_xor(a1, m, 32);
    a2 += __shfl_xor(a2, m, 32);
    a3 += __shfl_xor(a3, m, 32);
  }
  if (lane == 0) {
    float* o = out + (size_t)row * Rx;
    o[0] = a0; o[1] = a1; o[2] = a2; o[3] = a3;
  }
}

// ---------------------------------------------------------------------------
// Projection GEMM: out[row][col] = (sum_c X[row][c]*W[col][c] + LoRA) * scale + bias
// X: [ROWS, Cx] bf16, W: [Cx, Cx] bf16 (row = output feature)
// Wave computes 32(M) x 64(N); block = 4 waves = 128 rows x 64 cols.
// OUT_TRANS stores bf16 transposed per plane: out[(plane*Cx+col)*Sx + s]
// (used for V so the attention P*V operand becomes a row-major A-load).
// ---------------------------------------------------------------------------
template <int OUT_MODE /*0=f32, 1=bf16, 2=bf16 transposed*/, bool HAS_LORA, bool HAS_BIAS>
__global__ __launch_bounds__(128) void gemm_proj(const __bf16* __restrict__ X,
                                                 const __bf16* __restrict__ W,
                                                 const float* __restrict__ bias,
                                                 const float* __restrict__ lh,   // [ROWS, Rx]
                                                 const float* __restrict__ up,   // [3, Cx, Rx]
                                                 float scale,
                                                 __bf16* __restrict__ outb,
                                                 float* __restrict__ outf) {
  const int wave     = threadIdx.x >> 5;
  const int lane     = threadIdx.x & 31;
  const int row_base = blockIdx.x * 128 + wave * 32;
  const int col_base = blockIdx.y * 64;

  v8f acc[2][4];
#pragma unroll
  for (int mt = 0; mt < 2; ++mt)
#pragma unroll
    for (int nt = 0; nt < 4; ++nt) acc[mt][nt] = {};

  for (int k0 = 0; k0 < Cx; k0 += 32) {
    v16bf a0 = load_frag_a(X, Cx, row_base, k0);
    v16bf a1 = load_frag_a(X, Cx, row_base + 16, k0);
#pragma unroll
    for (int nt = 0; nt < 4; ++nt) {
      v16bf b = load_frag_b(W, Cx, col_base + nt * 16, k0);
      acc[0][nt] = wmma_bf16(a0, b, acc[0][nt]);
      acc[1][nt] = wmma_bf16(a1, b, acc[1][nt]);
    }
  }

  const int hi = lane >> 4, n = lane & 15;
  const int plane = row_base >> 10;
#pragma unroll
  for (int mt = 0; mt < 2; ++mt) {
#pragma unroll
    for (int nt = 0; nt < 4; ++nt) {
      const int col = col_base + nt * 16 + n;
      float u0 = 0.f, u1 = 0.f, u2 = 0.f, u3 = 0.f, bv = 0.f;
      if (HAS_LORA) {
        const float* u = up + ((size_t)plane * Cx + col) * Rx;
        u0 = u[0]; u1 = u[1]; u2 = u[2]; u3 = u[3];
      }
      if (HAS_BIAS) bv = bias[col];
#pragma unroll
      for (int j = 0; j < 8; ++j) {
        const int row = row_base + mt * 16 + hi * 8 + j;
        float v = acc[mt][nt][j];
        if (HAS_LORA) {
          const float* g = lh + (size_t)row * Rx;
          v += g[0] * u0 + g[1] * u1 + g[2] * u2 + g[3] * u3;
        }
        v = v * scale + bv;
        if (OUT_MODE == 0)      outf[(size_t)row * Cx + col] = v;
        else if (OUT_MODE == 1) outb[(size_t)row * Cx + col] = (__bf16)v;
        else {
          const int s = row & (Sx - 1);
          outb[((size_t)plane * Cx + col) * Sx + s] = (__bf16)v;
        }
      }
    }
  }
}

// ---------------------------------------------------------------------------
// Fused flash attention (transposed-score formulation).
// One wave per (plane, head, 16-query tile). Q pre-scaled by 1/sqrt(D).
// St = K_tile(16x64) * Q_tile^T  -> lane holds one query column's stats.
// O^T accumulated as V^T(64xK) * P(Kx16) via WMMA, f32.
// Vt is stored transposed: Vt[(plane*Cx + c)*Sx + s], so the V^T A-operand
// is a plain row-major fragment load (no scalar gathers).
// ---------------------------------------------------------------------------
__global__ __launch_bounds__(128) void attn_fused(const __bf16* __restrict__ Q,
                                                  const __bf16* __restrict__ Kk,
                                                  const __bf16* __restrict__ Vt,
                                                  __bf16* __restrict__ O) {
  const int wid  = blockIdx.x * 4 + (threadIdx.x >> 5);
  const int lane = threadIdx.x & 31;
  const int qt = wid & 63;            // 64 query tiles of 16
  const int h  = (wid >> 6) & 15;
  const int p  = wid >> 10;           // plane
  const int q0   = qt << 4;
  const int hcol = h << 6;
  const int hi   = lane >> 4;
  const int nl   = lane & 15;

  // Q^T as B-operands (lane = query column), loaded once
  const v16bf bq0 = load_frag_b(Q, Cx, p * Sx + q0, hcol);
  const v16bf bq1 = load_frag_b(Q, Cx, p * Sx + q0, hcol + 32);

  v8f oacc[4];
#pragma unroll
  for (int dt = 0; dt < 4; ++dt) oacc[dt] = {};
  float m_run = -1.0e30f, l_run = 0.f;

  for (int t0 = 0; t0 < Tx; t0 += 32) {
    const int kvp  = (p + (t0 >> 10)) % 3;           // triplane permuted concat
    const int kr   = t0 & (Sx - 1);
    const int krow = kvp * Sx + kr;

    // transposed scores: rows = keys, cols = queries
    v8f s0 = {}, s1 = {};
    s0 = wmma_bf16(load_frag_a(Kk, Cx, krow,      hcol     ), bq0, s0);
    s0 = wmma_bf16(load_frag_a(Kk, Cx, krow,      hcol + 32), bq1, s0);
    s1 = wmma_bf16(load_frag_a(Kk, Cx, krow + 16, hcol     ), bq0, s1);
    s1 = wmma_bf16(load_frag_a(Kk, Cx, krow + 16, hcol + 32), bq1, s1);

    // per-query (per-lane) online softmax stats over 32 keys
    float mx = -1.0e30f;
#pragma unroll
    for (int j = 0; j < 8; ++j) { mx = fmaxf(mx, s0[j]); mx = fmaxf(mx, s1[j]); }
    mx = fmaxf(mx, __shfl_xor(mx, 16, 32));
    const float m_new = fmaxf(m_run, mx);
    const float corr  = __expf(m_run - m_new);

    float E0[8], E1[8];
    float ls = 0.f;
#pragma unroll
    for (int j = 0; j < 8; ++j) {
      E0[j] = __expf(s0[j] - m_new);
      E1[j] = __expf(s1[j] - m_new);
      ls += E0[j] + E1[j];
    }
    ls += __shfl_xor(ls, 16, 32);
    l_run = l_run * corr + ls;
    m_run = m_new;

    // Build P B-operand (32 keys x 16 queries). B element e of lane L needs
    // key k = e + 16*hi at column nl. Local regs hold keys j+8*hi of each
    // 16-key tile, so exchange halves via lane-xor-16.
    v16bf bp;
#pragma unroll
    for (int j = 0; j < 8; ++j) {
      float c0 = __shfl_xor(E0[j], 16, 32);
      float c1 = __shfl_xor(E1[j], 16, 32);
      float e_lo = (hi == 0) ? E0[j] : c1;    // keys 0..7   | 16..23
      float e_hi = (hi == 0) ? c0    : E1[j]; // keys 8..15  | 24..31
      bp[j]     = (__bf16)e_lo;
      bp[j + 8] = (__bf16)e_hi;
    }

    // O^T += V^T * P ; V^T tile is a row-major A-load from transposed V
    const __bf16* vplane = Vt + (size_t)kvp * Cx * Sx;
#pragma unroll
    for (int dt = 0; dt < 4; ++dt) {
#pragma unroll
      for (int j = 0; j < 8; ++j) oacc[dt][j] *= corr;
      v16bf av = load_frag_a(vplane, Sx, hcol + dt * 16, kr);
      oacc[dt] = wmma_bf16(av, bp, oacc[dt]);
    }
  }

  const float inv  = 1.f / l_run;
  const int   orow = p * Sx + q0 + nl;   // lane = query column
#pragma unroll
  for (int dt = 0; dt < 4; ++dt) {
#pragma unroll
    for (int j = 0; j < 8; ++j) {
      O[(size_t)orow * Cx + hcol + dt * 16 + hi * 8 + j] = (__bf16)(oacc[dt][j] * inv);
    }
  }
}

// ---------------------------------------------------------------------------
// h[row][col] += sum_r g[row][r] * up[plane][col][r]
// ---------------------------------------------------------------------------
__global__ __launch_bounds__(256) void lora_up_add_k(float* __restrict__ hbuf,
                                                     const float* __restrict__ g,
                                                     const float* __restrict__ up, int n) {
  int i = blockIdx.x * blockDim.x + threadIdx.x;
  if (i >= n) return;
  const int row = i >> 10;
  const int col = i & (Cx - 1);
  const int p   = row >> 10;
  const float* gr = g + (size_t)row * Rx;
  const float* u  = up + ((size_t)p * Cx + col) * Rx;
  hbuf[i] += gr[0] * u[0] + gr[1] * u[1] + gr[2] * u[2] + gr[3] * u[3];
}

// ---------------------------------------------------------------------------
// Host launcher
// ---------------------------------------------------------------------------
extern "C" void kernel_launch(void* const* d_in, const int* in_sizes, int n_in,
                              void* d_out, int out_size, void* d_ws, size_t ws_size,
                              hipStream_t stream) {
  const float* x_f  = (const float*)d_in[0];
  const float* Wq_f = (const float*)d_in[1];
  const float* Wk_f = (const float*)d_in[2];
  const float* Wv_f = (const float*)d_in[3];
  const float* Wo_f = (const float*)d_in[4];
  const float* b_o  = (const float*)d_in[5];
  const float* q_dn = (const float*)d_in[6];
  const float* q_up = (const float*)d_in[7];
  const float* k_dn = (const float*)d_in[8];
  const float* k_up = (const float*)d_in[9];
  const float* v_dn = (const float*)d_in[10];
  const float* v_up = (const float*)d_in[11];
  const float* o_dn = (const float*)d_in[12];
  const float* o_up = (const float*)d_in[13];
  float* out = (float*)d_out;

  // workspace carve-up
  char* w = (char*)d_ws;
  size_t off = 0;
  auto carve = [&](size_t bytes) -> void* {
    void* p = (void*)(w + off);
    off += (bytes + 255) & ~(size_t)255;
    return p;
  };
  const size_t XN = (size_t)ROWS * Cx;     // 3,145,728
  const size_t WN = (size_t)Cx * Cx;       // 1,048,576
  __bf16* xb  = (__bf16*)carve(XN * 2);
  __bf16* Wqb = (__bf16*)carve(WN * 2);
  __bf16* Wkb = (__bf16*)carve(WN * 2);
  __bf16* Wvb = (__bf16*)carve(WN * 2);
  __bf16* Wob = (__bf16*)carve(WN * 2);
  __bf16* Qb  = (__bf16*)carve(XN * 2);
  __bf16* Kb  = (__bf16*)carve(XN * 2);
  __bf16* Vtb = (__bf16*)carve(XN * 2);    // V, transposed per plane
  __bf16* Ab  = (__bf16*)carve(XN * 2);    // attention output, bf16
  float*  hq  = (float*)carve((size_t)ROWS * Rx * 4);
  float*  hk  = (float*)carve((size_t)ROWS * Rx * 4);
  float*  hv  = (float*)carve((size_t)ROWS * Rx * 4);
  float*  go  = (float*)carve((size_t)ROWS * Rx * 4);

  // 1) fp32 -> bf16 conversions
  cvt_f32_bf16<<<(int)(XN / 4 + 255) / 256, 256, 0, stream>>>(x_f, xb, (int)(XN / 4));
  cvt_f32_bf16<<<(int)(WN / 4 + 255) / 256, 256, 0, stream>>>(Wq_f, Wqb, (int)(WN / 4));
  cvt_f32_bf16<<<(int)(WN / 4 + 255) / 256, 256, 0, stream>>>(Wk_f, Wkb, (int)(WN / 4));
  cvt_f32_bf16<<<(int)(WN / 4 + 255) / 256, 256, 0, stream>>>(Wv_f, Wvb, (int)(WN / 4));
  cvt_f32_bf16<<<(int)(WN / 4 + 255) / 256, 256, 0, stream>>>(Wo_f, Wob, (int)(WN / 4));

  // 2) LoRA down-projections of x (fp32, rank 4)
  lora_down_k<<<ROWS / 4, 128, 0, stream>>>(x_f, q_dn, hq);
  lora_down_k<<<ROWS / 4, 128, 0, stream>>>(x_f, k_dn, hk);
  lora_down_k<<<ROWS / 4, 128, 0, stream>>>(x_f, v_dn, hv);

  // 3) Q/K/V projections (WMMA) + LoRA up; fold 1/sqrt(D)=0.125 into Q;
  //    V stored transposed for the attention P*V stage
  dim3 ggrid(ROWS / 128, Cx / 64);
  gemm_proj<1, true, false><<<ggrid, 128, 0, stream>>>(xb, Wqb, nullptr, hq, q_up, 0.125f, Qb, nullptr);
  gemm_proj<1, true, false><<<ggrid, 128, 0, stream>>>(xb, Wkb, nullptr, hk, k_up, 1.0f,   Kb, nullptr);
  gemm_proj<2, true, false><<<ggrid, 128, 0, stream>>>(xb, Wvb, nullptr, hv, v_up, 1.0f,   Vtb, nullptr);

  // 4) fused triplane flash attention: 3 planes * 16 heads * 64 q-tiles waves
  attn_fused<<<(3 * Hx * (Sx / 16)) / 4, 128, 0, stream>>>(Qb, Kb, Vtb, Ab);

  // 5) output projection (WMMA) + bias -> fp32 h in d_out
  gemm_proj<0, false, true><<<ggrid, 128, 0, stream>>>(Ab, Wob, b_o, nullptr, nullptr, 1.0f, nullptr, out);

  // 6) O-LoRA applied to h (matches reference: lora of the projected output)
  lora_down_k<<<ROWS / 4, 128, 0, stream>>>(out, o_dn, go);
  lora_up_add_k<<<(int)(XN + 255) / 256, 256, 0, stream>>>(out, go, o_up, (int)XN);
}
